// TransferModel_67327907332773
// MI455X (gfx1250) — matrix operand order repaired and checked by reading, not compiled, not run on hardware
//
#include <hip/hip_runtime.h>

// ---------------------------------------------------------------------------
// Cluster-wise whitening-coloring style transfer for MI455X (gfx1250).
// eigh is replaced by Denman-Beavers Newton-Schulz iteration so that all
// heavy math is bf16 WMMA (v_wmma_f32_16x16x32_bf16) with f32 accumulation.
// Row-major A tiles are fed to LDS via the Tensor Data Mover (TDM).
// ---------------------------------------------------------------------------

#define B_ 8
#define C_ 512
#define N_ 4096          // H*W
#define K_ 3
#define S_ 2             // content, style
#define MATS 48          // S_*B_*K_
#define CC (C_ * C_)
#define ALPHA_ 0.6f
#define GAMMA_ 0.1f
#define KM_ITERS 10
#define NS_ITERS 12

typedef __bf16 bf16;
typedef __attribute__((ext_vector_type(16))) __bf16 v16bf;
typedef __attribute__((ext_vector_type(8)))  __bf16 v8bf;
typedef __attribute__((ext_vector_type(8)))  float  v8f;
typedef __attribute__((ext_vector_type(4)))  unsigned int v4u;
typedef __attribute__((ext_vector_type(8)))  int v8i;
typedef __attribute__((ext_vector_type(4)))  int v4i;

#define BM 64            // tile rows
#define BN 128           // tile cols
#define BK 32            // K step
#define LDA 40           // 32 + 8 halves pad; multiple of 8 keeps 16B alignment

// Assemble a 16-half fragment from two aligned 8-half LDS chunks (2x ds_load_b128).
__device__ __forceinline__ v16bf frag_from2(const bf16* p0, const bf16* p1) {
  v8bf lo = *(const v8bf*)p0;
  v8bf hi = *(const v8bf*)p1;
  v16bf r;
#pragma unroll
  for (int i = 0; i < 8; ++i) { r[i] = lo[i]; r[i + 8] = hi[i]; }
  return r;
}

// TDM: load a BM x BK bf16 tile from a row-major matrix (row stride C_ elems)
// into LDS with 16B padding after each 64B row (LDS row stride = LDA halves).
// D# built per CDNA5 ISA 8.3/8.4 (2D tile, data_size=2B, pad 16DW->+4DW).
// This toolchain exposes the 6-arg builtin form (g0, g1, g2, g3, g4, cpol).
__device__ __forceinline__ void tdm_load_A_tile(const bf16* gsrc, unsigned lds_byte) {
  const unsigned long long ga = (unsigned long long)(uintptr_t)gsrc;
  v4u g0;
  g0[0] = 1u;                                   // count=1, user descriptor
  g0[1] = lds_byte;                             // lds_addr
  g0[2] = (unsigned)ga;                         // global_addr[31:0]
  g0[3] = (unsigned)((ga >> 32) & 0x01FFFFFFu)  // global_addr[56:32]
          | (2u << 30);                         // type=2 (image)
  v8i g1;
  g1[0] = (1 << 16)    // data_size = 2 bytes
        | (1 << 20)    // pad_enable
        | (3 << 22)    // pad_interval: 16 DWORDs (one 64B tile row)
        | (3 << 25);   // pad_amount: 4 DWORDs (16B -> 80B LDS row stride)
  g1[1] = (int)((C_ & 0xFFFF) << 16);                 // tensor_dim0 lo16
  g1[2] = (int)((C_ >> 16) | ((C_ & 0xFFFF) << 16));  // tensor_dim0 hi, dim1 lo
  g1[3] = (int)((C_ >> 16) | (BK << 16));             // dim1 hi, tile_dim0=BK
  g1[4] = BM;                                         // tile_dim1=BM, tile_dim2=0
  g1[5] = C_;                                         // tensor_dim0_stride lo32
  g1[6] = (int)((CC & 0xFFFF) << 16);                 // stride0 hi, stride1 lo
  g1[7] = (int)(CC >> 16);                            // stride1 hi
  v4i z4 = {0, 0, 0, 0};
  v8i z8 = {0, 0, 0, 0, 0, 0, 0, 0};
  __builtin_amdgcn_tensor_load_to_lds(g0, g1, z4, z4, z8, 0);
}

// 64x128 output tile GEMM core: 256 threads = 8 wave32s in 2x4; each wave owns
// a 32x32 accumulator (4 subtiles) -> 4 v_wmma_f32_16x16x32_bf16 per K step.
// LoadA fills As[m][k] (BM x BK), LoadB fills Bs[n][k] (BN x BK, k-major per
// column), WaitF drains async engines, Epi consumes f32 results.
template <typename LoadA, typename LoadB, typename WaitF, typename Epi>
__device__ __forceinline__ void gemm_tile(int Ksize, LoadA loadA, LoadB loadB,
                                          WaitF waitf, Epi epi) {
  __shared__ __align__(16) bf16 As[BM * LDA];
  __shared__ __align__(16) bf16 Bs[BN * LDA];
  const int tid  = threadIdx.x;
  const int wv   = tid >> 5;
  const int lane = tid & 31;
  const int rr   = (wv & 1) * 32;   // wave row base (0 / 32)
  const int cp   = (wv >> 1) * 32;  // wave col base (0..96)
  v8f acc00 = {}, acc01 = {}, acc10 = {}, acc11 = {};
  for (int kk = 0; kk < Ksize; kk += BK) {
    loadA(As, kk, tid);
    loadB(Bs, kk, tid);
    waitf();
    __syncthreads();
    // A fragment: lane<16 -> K {0..7,16..23}; lane>=16 -> K {8..15,24..31}
    const int klo = (lane < 16) ? 0 : 8;
    const int kb  = (lane < 16) ? 0 : 16;
    const int m0  = rr + (lane & 15);
    v16bf a0 = frag_from2(&As[m0 * LDA + klo], &As[m0 * LDA + klo + 16]);
    v16bf a1 = frag_from2(&As[(m0 + 16) * LDA + klo], &As[(m0 + 16) * LDA + klo + 16]);
    // B fragment: lane<16 -> K {0..7,16..23} of column lane; lane>=16 -> K {8..15,24..31}
    const int n0 = cp + (lane & 15);
    v16bf b0 = frag_from2(&Bs[n0 * LDA + klo], &Bs[n0 * LDA + klo + 16]);
    v16bf b1 = frag_from2(&Bs[(n0 + 16) * LDA + klo], &Bs[(n0 + 16) * LDA + klo + 16]);
    acc00 = __builtin_amdgcn_wmma_f32_16x16x32_bf16(false, a0, false, b0, (short)0, acc00, false, false);
    acc01 = __builtin_amdgcn_wmma_f32_16x16x32_bf16(false, a0, false, b1, (short)0, acc01, false, false);
    acc10 = __builtin_amdgcn_wmma_f32_16x16x32_bf16(false, a1, false, b0, (short)0, acc10, false, false);
    acc11 = __builtin_amdgcn_wmma_f32_16x16x32_bf16(false, a1, false, b1, (short)0, acc11, false, false);
    __syncthreads();
  }
#pragma unroll
  for (int i = 0; i < 8; ++i) {
    const int r0 = rr + ((lane < 16) ? i : i + 8);
    const int r1 = r0 + 16;
    const int c0 = cp + (lane & 15);
    epi(r0, c0, acc00[i]);
    epi(r0, c0 + 16, acc01[i]);
    epi(r1, c0, acc10[i]);
    epi(r1, c0 + 16, acc11[i]);
  }
}

__device__ __forceinline__ void no_wait() {}

// ---------------------------------------------------------------------------
// K-means: one 256-thread block per (set, batch). Two-pass accumulation with
// LDS assignment cache (no per-element LDS atomics).
// ---------------------------------------------------------------------------
__global__ void kmeans_kernel(const float* __restrict__ fc,
                              const float* __restrict__ fs,
                              int* __restrict__ assign) {
  const int wg = blockIdx.x;          // 0..15
  const int s  = wg >> 3;
  const int b  = wg & 7;
  const float* f = (s == 0 ? fc : fs) + (size_t)b * C_ * N_;
  int* asg = assign + ((size_t)s * B_ + b) * N_;

  __shared__ float cent[K_][C_];
  __shared__ float sums[K_][C_];
  __shared__ float cnt[K_];
  __shared__ float ccsq[K_];
  __shared__ unsigned char abuf[N_];
  const int tid = threadIdx.x;

  const int initn[K_] = {0, 2047, 4095};  // linspace(0, N-1, 3) -> int32
  for (int i = tid; i < K_ * C_; i += 256) {
    const int k = i / C_, c = i % C_;
    cent[k][c] = f[(size_t)c * N_ + initn[k]];
  }
  __syncthreads();

  for (int it = 0; it <= KM_ITERS; ++it) {
    const bool final_pass = (it == KM_ITERS);
    if (tid < K_) { cnt[tid] = 0.f; ccsq[tid] = 0.f; }
    __syncthreads();
    for (int i = tid; i < K_ * C_; i += 256) {
      const int k = i / C_, c = i % C_;
      const float v = cent[k][c];
      atomicAdd(&ccsq[k], v * v);
    }
    __syncthreads();

    // Pass 1: assignments.
    for (int n = tid; n < N_; n += 256) {
      float d0 = 0.f, d1 = 0.f, d2 = 0.f;
      for (int c = 0; c < C_; ++c) {
        const float v = f[(size_t)c * N_ + n];
        d0 += v * cent[0][c];
        d1 += v * cent[1][c];
        d2 += v * cent[2][c];
      }
      const float s0 = ccsq[0] - 2.f * d0;
      const float s1 = ccsq[1] - 2.f * d1;
      const float s2 = ccsq[2] - 2.f * d2;
      int best = 0; float bs = s0;
      if (s1 < bs) { bs = s1; best = 1; }
      if (s2 < bs) { bs = s2; best = 2; }
      if (final_pass) asg[n] = best;
      else            abuf[n] = (unsigned char)best;
    }
    __syncthreads();
    if (final_pass) break;

    // Pass 2: counts (few atomics) + channel-parallel sums (no atomics).
    {
      float c0 = 0.f, c1 = 0.f, c2 = 0.f;
      for (int n = tid; n < N_; n += 256) {
        const int a = abuf[n];
        c0 += (a == 0) ? 1.f : 0.f;
        c1 += (a == 1) ? 1.f : 0.f;
        c2 += (a == 2) ? 1.f : 0.f;
      }
      atomicAdd(&cnt[0], c0);
      atomicAdd(&cnt[1], c1);
      atomicAdd(&cnt[2], c2);
    }
    for (int c = tid; c < C_; c += 256) {
      float s0 = 0.f, s1 = 0.f, s2 = 0.f;
      const float* row = f + (size_t)c * N_;
      for (int n = 0; n < N_; ++n) {
        const float v = row[n];
        const int a = abuf[n];
        s0 += (a == 0) ? v : 0.f;
        s1 += (a == 1) ? v : 0.f;
        s2 += (a == 2) ? v : 0.f;
      }
      sums[0][c] = s0; sums[1][c] = s1; sums[2][c] = s2;
    }
    __syncthreads();
    for (int i = tid; i < K_ * C_; i += 256) {
      const int k = i / C_, c = i % C_;
      if (cnt[k] > 0.f) cent[k][c] = sums[k][c] / fmaxf(cnt[k], 1.f);
    }
    __syncthreads();
  }
}

// ---------------------------------------------------------------------------
// Per-(set,batch,cluster) counts and means.
// ---------------------------------------------------------------------------
__global__ void mean_kernel(const float* __restrict__ fc,
                            const float* __restrict__ fs,
                            const int* __restrict__ assign,
                            float* __restrict__ mu,
                            float* __restrict__ counts) {
  const int mz = blockIdx.x;  // 0..47
  const int s = mz / (B_ * K_);
  const int bk = mz % (B_ * K_);
  const int b = bk / K_, k = bk % K_;
  const float* f = (s == 0 ? fc : fs) + (size_t)b * C_ * N_;
  const int* asg = assign + ((size_t)s * B_ + b) * N_;
  const int tid = threadIdx.x;

  __shared__ float red[256];
  __shared__ float cnt_sh;
  float c0 = 0.f;
  for (int n = tid; n < N_; n += 256) c0 += (asg[n] == k) ? 1.f : 0.f;
  red[tid] = c0;
  __syncthreads();
  for (int st = 128; st > 0; st >>= 1) {
    if (tid < st) red[tid] += red[tid + st];
    __syncthreads();
  }
  if (tid == 0) { cnt_sh = red[0]; counts[mz] = red[0]; }
  __syncthreads();
  const float inv = 1.f / fmaxf(cnt_sh, 1.f);
  for (int c = tid; c < C_; c += 256) {
    float acc = 0.f;
    const float* row = f + (size_t)c * N_;
    for (int n = 0; n < N_; ++n) acc += (asg[n] == k) ? row[n] : 0.f;
    mu[(size_t)mz * C_ + c] = acc * inv;
  }
}

// ---------------------------------------------------------------------------
// Covariance: D = X X^T / max(n-1,1) + gamma I, X = (f - mu) * mask.
// Writes bf16 into Y0 (unnormalized NS seed) and accumulates trace.
// ---------------------------------------------------------------------------
__global__ void cov_kernel(const float* __restrict__ fc,
                           const float* __restrict__ fs,
                           const int* __restrict__ assign,
                           const float* __restrict__ mu,
                           const float* __restrict__ counts,
                           bf16* __restrict__ Y0,
                           float* __restrict__ trace) {
  const int mz = blockIdx.z;  // 0..47
  const int s = mz / (B_ * K_);
  const int bk = mz % (B_ * K_);
  const int b = bk / K_, k = bk % K_;
  const float* f = (s == 0 ? fc : fs) + (size_t)b * C_ * N_;
  const int* asg = assign + ((size_t)s * B_ + b) * N_;
  const float* mup = mu + (size_t)mz * C_;
  const int row0 = blockIdx.y * BM;
  const int col0 = blockIdx.x * BN;
  const float cnt = counts[mz];
  const float scl = 1.f / fmaxf(cnt - 1.f, 1.f);
  bf16* D = Y0 + (size_t)mz * CC;

  auto loadA = [&](bf16* As, int kk, int tid) {
    for (int idx = tid; idx < BM * BK; idx += 256) {
      const int n = idx & 31, c = idx >> 5;
      const float msk = (asg[kk + n] == k) ? 1.f : 0.f;
      const float v = (f[(size_t)(row0 + c) * N_ + kk + n] - mup[row0 + c]) * msk;
      As[c * LDA + n] = (bf16)v;
    }
  };
  auto loadB = [&](bf16* Bs, int kk, int tid) {
    for (int idx = tid; idx < BN * BK; idx += 256) {
      const int n = idx & 31, c = idx >> 5;   // c: 0..127
      const float msk = (asg[kk + n] == k) ? 1.f : 0.f;
      const float v = (f[(size_t)(col0 + c) * N_ + kk + n] - mup[col0 + c]) * msk;
      Bs[c * LDA + n] = (bf16)v;
    }
  };

  float diag = 0.f;
  auto epi = [&](int lr, int lc, float v) {
    const int gr = row0 + lr, gc = col0 + lc;
    const float d = v * scl + ((gr == gc) ? GAMMA_ : 0.f);
    D[(size_t)gr * C_ + gc] = (bf16)d;
    if (gr == gc) diag += d;
  };
  gemm_tile(N_, loadA, loadB, no_wait, epi);

  __shared__ float red[256];
  red[threadIdx.x] = diag;
  __syncthreads();
  for (int st = 128; st > 0; st >>= 1) {
    if (threadIdx.x < st) red[threadIdx.x] += red[threadIdx.x + st];
    __syncthreads();
  }
  if (threadIdx.x == 0 && red[0] != 0.f) atomicAdd(&trace[mz], red[0]);
}

// ---------------------------------------------------------------------------
// Newton-Schulz init: Y0 = cov / trace (trace >= lambda_max for PSD), Z0 = I.
// ---------------------------------------------------------------------------
__global__ void ns_init_kernel(bf16* __restrict__ Y, bf16* __restrict__ Z,
                               const float* __restrict__ trace) {
  const int mz = blockIdx.y;
  const size_t base = (size_t)mz * CC;
  const float inv = 1.f / fmaxf(trace[mz], 1e-12f);
  const int i = blockIdx.x * 256 + threadIdx.x;  // grid.x = CC/256
  const int r = i >> 9, c = i & 511;
  Y[base + i] = (bf16)((float)Y[base + i] * inv);
  Z[base + i] = (r == c) ? (bf16)1.0f : (bf16)0.0f;
}

// T = 1.5*I - 0.5*(Z @ Y)
__global__ void ns_t_kernel(const bf16* __restrict__ Zc,
                            const bf16* __restrict__ Yc,
                            bf16* __restrict__ T) {
  const int mz = blockIdx.z;
  const size_t off = (size_t)mz * CC;
  const bf16* A = Zc + off;
  const bf16* Bm = Yc + off;
  bf16* D = T + off;
  const int row0 = blockIdx.y * BM, col0 = blockIdx.x * BN;

  auto loadA = [&](bf16* As, int kk, int tid) {
    if (tid == 0)
      tdm_load_A_tile(A + (size_t)row0 * C_ + kk, (unsigned)(uintptr_t)As);
  };
  auto loadB = [&](bf16* Bs, int kk, int tid) {
    for (int idx = tid; idx < BN * BK; idx += 256) {
      const int n = idx & 127, kr = idx >> 7;
      Bs[n * LDA + kr] = Bm[(size_t)(kk + kr) * C_ + col0 + n];
    }
  };
  auto waitf = [&]() {
    if (threadIdx.x == 0) __builtin_amdgcn_s_wait_tensorcnt(0);
  };
  auto epi = [&](int lr, int lc, float v) {
    const int gr = row0 + lr, gc = col0 + lc;
    const float d = ((gr == gc) ? 1.5f : 0.f) - 0.5f * v;
    D[(size_t)gr * C_ + gc] = (bf16)d;
  };
  gemm_tile(C_, loadA, loadB, waitf, epi);
}

// z < MATS: Ynext = Y @ T ; z >= MATS: Znext = T @ Z
__global__ void ns_yz_kernel(const bf16* __restrict__ Yc,
                             const bf16* __restrict__ Zc,
                             const bf16* __restrict__ T,
                             bf16* __restrict__ Yn,
                             bf16* __restrict__ Zn) {
  const int z = blockIdx.z;
  const int mz = (z < MATS) ? z : z - MATS;
  const bool doZ = (z >= MATS);
  const size_t off = (size_t)mz * CC;
  const bf16* A = doZ ? (T + off) : (Yc + off);
  const bf16* Bm = doZ ? (Zc + off) : (T + off);
  bf16* D = doZ ? (Zn + off) : (Yn + off);
  const int row0 = blockIdx.y * BM, col0 = blockIdx.x * BN;

  auto loadA = [&](bf16* As, int kk, int tid) {
    if (tid == 0)
      tdm_load_A_tile(A + (size_t)row0 * C_ + kk, (unsigned)(uintptr_t)As);
  };
  auto loadB = [&](bf16* Bs, int kk, int tid) {
    for (int idx = tid; idx < BN * BK; idx += 256) {
      const int n = idx & 127, kr = idx >> 7;
      Bs[n * LDA + kr] = Bm[(size_t)(kk + kr) * C_ + col0 + n];
    }
  };
  auto waitf = [&]() {
    if (threadIdx.x == 0) __builtin_amdgcn_s_wait_tensorcnt(0);
  };
  auto epi = [&](int lr, int lc, float v) {
    D[(size_t)(row0 + lr) * C_ + col0 + lc] = (bf16)v;
  };
  gemm_tile(C_, loadA, loadB, waitf, epi);
}

// M[b,k] = sqrt(ts/tc) * Y_style @ Z_content  (= cov_s^{1/2} @ cov_c^{-1/2})
__global__ void m_kernel(const bf16* __restrict__ Yf,
                         const bf16* __restrict__ Zf,
                         const float* __restrict__ trace,
                         bf16* __restrict__ M) {
  const int mz = blockIdx.z;           // 0..23 == b*K_+k
  const int cIdx = mz;                 // content matrices occupy [0,24)
  const int sIdx = B_ * K_ + mz;       // style matrices occupy [24,48)
  const bf16* A = Yf + (size_t)sIdx * CC;
  const bf16* Bm = Zf + (size_t)cIdx * CC;
  bf16* D = M + (size_t)mz * CC;
  const float scl = sqrtf(fmaxf(trace[sIdx], 1e-12f) / fmaxf(trace[cIdx], 1e-12f));
  const int row0 = blockIdx.y * BM, col0 = blockIdx.x * BN;

  auto loadA = [&](bf16* As, int kk, int tid) {
    if (tid == 0)
      tdm_load_A_tile(A + (size_t)row0 * C_ + kk, (unsigned)(uintptr_t)As);
  };
  auto loadB = [&](bf16* Bs, int kk, int tid) {
    for (int idx = tid; idx < BN * BK; idx += 256) {
      const int n = idx & 127, kr = idx >> 7;
      Bs[n * LDA + kr] = Bm[(size_t)(kk + kr) * C_ + col0 + n];
    }
  };
  auto waitf = [&]() {
    if (threadIdx.x == 0) __builtin_amdgcn_s_wait_tensorcnt(0);
  };
  auto epi = [&](int lr, int lc, float v) {
    D[(size_t)(row0 + lr) * C_ + col0 + lc] = (bf16)(v * scl);
  };
  gemm_tile(C_, loadA, loadB, waitf, epi);
}

// v[b,k] = M[b,k] @ mu_content[b,k]
__global__ void v_kernel(const bf16* __restrict__ M,
                         const float* __restrict__ mu,
                         float* __restrict__ vvec) {
  const int mz = blockIdx.x;  // 0..23
  const bf16* Mm = M + (size_t)mz * CC;
  const float* muc = mu + (size_t)mz * C_;  // content means occupy [0,24)*C_
  float* vv = vvec + (size_t)mz * C_;
  for (int c = threadIdx.x; c < C_; c += 256) {
    float acc = 0.f;
    for (int j = 0; j < C_; ++j) acc += (float)Mm[(size_t)c * C_ + j] * muc[j];
    vv[c] = acc;
  }
}

// out = alpha*(M@fc - v + mu_s) + (1-alpha)*fc, scattered by content cluster.
__global__ void apply_kernel(const bf16* __restrict__ M,
                             const float* __restrict__ fc,
                             const int* __restrict__ assign,
                             const float* __restrict__ mu,
                             const float* __restrict__ vvec,
                             float* __restrict__ out) {
  const int mz = blockIdx.z;  // 0..23 == b*K_+k
  const int b = mz / K_, k = mz % K_;
  const bf16* A = M + (size_t)mz * CC;
  const float* fcb = fc + (size_t)b * C_ * N_;
  const int* asg = assign + (size_t)b * N_;               // content set
  const float* mus = mu + ((size_t)(B_ * K_) + mz) * C_;  // style means
  const float* vv = vvec + (size_t)mz * C_;
  float* ob = out + (size_t)b * C_ * N_;
  const int row0 = blockIdx.y * BM;   // channel
  const int col0 = blockIdx.x * BN;   // pixel

  auto loadA = [&](bf16* As, int kk, int tid) {
    if (tid == 0)
      tdm_load_A_tile(A + (size_t)row0 * C_ + kk, (unsigned)(uintptr_t)As);
  };
  auto loadB = [&](bf16* Bs, int kk, int tid) {
    for (int idx = tid; idx < BN * BK; idx += 256) {
      const int n = idx & 127, kr = idx >> 7;
      Bs[n * LDA + kr] = (bf16)fcb[(size_t)(kk + kr) * N_ + col0 + n];
    }
    if (kk + BK < C_)
      __builtin_prefetch(&fcb[(size_t)(kk + BK) * N_ + col0 + (tid & 127)], 0, 1);
  };
  auto waitf = [&]() {
    if (threadIdx.x == 0) __builtin_amdgcn_s_wait_tensorcnt(0);
  };
  auto epi = [&](int lr, int lc, float v) {
    const int gr = row0 + lr;   // channel
    const int gc = col0 + lc;   // pixel
    if (asg[gc] == k) {
      const float fv = fcb[(size_t)gr * N_ + gc];
      ob[(size_t)gr * N_ + gc] =
          ALPHA_ * (v - vv[gr] + mus[gr]) + (1.f - ALPHA_) * fv;
    }
  };
  gemm_tile(C_, loadA, loadB, waitf, epi);
}

// ---------------------------------------------------------------------------
extern "C" void kernel_launch(void* const* d_in, const int* in_sizes, int n_in,
                              void* d_out, int out_size, void* d_ws, size_t ws_size,
                              hipStream_t stream) {
  const float* fc = (const float*)d_in[0];
  const float* fs = (const float*)d_in[1];
  float* outp = (float*)d_out;

  char* w = (char*)d_ws;
  auto carve = [&](size_t bytes) -> char* {
    char* p = w;
    w += (bytes + 255) & ~(size_t)255;
    return p;
  };
  int*   assign = (int*)  carve((size_t)S_ * B_ * N_ * 4);
  float* counts = (float*)carve((size_t)MATS * 4);
  float* mu     = (float*)carve((size_t)MATS * C_ * 4);
  float* trace  = (float*)carve((size_t)MATS * 4);
  float* vvec   = (float*)carve((size_t)B_ * K_ * C_ * 4);
  bf16*  Ybuf   = (bf16*) carve((size_t)2 * MATS * CC * 2);
  bf16*  Zbuf   = (bf16*) carve((size_t)2 * MATS * CC * 2);
  bf16*  Tbuf   = (bf16*) carve((size_t)MATS * CC * 2);
  bf16*  Mmat   = (bf16*) carve((size_t)B_ * K_ * CC * 2);

  (void)hipMemsetAsync(trace, 0, MATS * sizeof(float), stream);

  kmeans_kernel<<<S_ * B_, 256, 0, stream>>>(fc, fs, assign);
  mean_kernel<<<MATS, 256, 0, stream>>>(fc, fs, assign, mu, counts);

  bf16* Y0 = Ybuf;  // seed: covariance written here by cov_kernel
  bf16* Z0 = Zbuf;
  cov_kernel<<<dim3(C_ / BN, C_ / BM, MATS), 256, 0, stream>>>(
      fc, fs, assign, mu, counts, Y0, trace);
  ns_init_kernel<<<dim3(CC / 256, MATS), 256, 0, stream>>>(Y0, Z0, trace);

  const size_t mcc = (size_t)MATS * CC;
  int p = 0;
  for (int it = 0; it < NS_ITERS; ++it) {
    bf16* Yc = Ybuf + (size_t)p * mcc;
    bf16* Zc = Zbuf + (size_t)p * mcc;
    bf16* Yn = Ybuf + (size_t)(1 - p) * mcc;
    bf16* Zn = Zbuf + (size_t)(1 - p) * mcc;
    ns_t_kernel<<<dim3(C_ / BN, C_ / BM, MATS), 256, 0, stream>>>(Zc, Yc, Tbuf);
    ns_yz_kernel<<<dim3(C_ / BN, C_ / BM, 2 * MATS), 256, 0, stream>>>(
        Yc, Zc, Tbuf, Yn, Zn);
    p ^= 1;
  }
  bf16* Yf = Ybuf + (size_t)p * mcc;
  bf16* Zf = Zbuf + (size_t)p * mcc;

  m_kernel<<<dim3(C_ / BN, C_ / BM, B_ * K_), 256, 0, stream>>>(Yf, Zf, trace, Mmat);
  v_kernel<<<B_ * K_, 256, 0, stream>>>(Mmat, mu, vvec);
  apply_kernel<<<dim3(N_ / BN, C_ / BM, B_ * K_), 256, 0, stream>>>(
      Mmat, fc, assign, mu, vvec, outp);
}
